// VQ_VAE_9620726743267
// MI455X (gfx1250) — matrix operand order repaired and checked by reading, not compiled, not run on hardware
//
#include <hip/hip_runtime.h>
#include <math.h>

// ---------------------------------------------------------------------------
// Types for CDNA5 WMMA / TDM
// ---------------------------------------------------------------------------
typedef __attribute__((ext_vector_type(16))) __bf16        v16bf;
typedef __attribute__((ext_vector_type(8)))  float         v8f;
typedef __attribute__((ext_vector_type(2)))  float         v2f;
typedef __attribute__((ext_vector_type(8)))  unsigned int  v8u;
typedef __attribute__((ext_vector_type(4)))  unsigned int  v4u;

__device__ __forceinline__ unsigned short f2bf(float f) {
  unsigned int u = __float_as_uint(f);
  u += 0x7FFFu + ((u >> 16) & 1u);           // round-to-nearest-even
  return (unsigned short)(u >> 16);
}
__device__ __forceinline__ float bf2f(unsigned short h) {
  return __uint_as_float(((unsigned int)h) << 16);
}
// ReLU on a packed pair of bf16 (clear negative halves)
__device__ __forceinline__ unsigned int relu_pair(unsigned int u) {
  if (u & 0x00008000u) u &= 0xFFFF0000u;
  if (u & 0x80000000u) u &= 0x0000FFFFu;
  return u;
}

// ---------------------------------------------------------------------------
// Spatial implicit-GEMM conv, bf16 WMMA 16x16x32, NHWC activations.
// Fully templated on shape: all address math is shifts/constant mults.
// Block = 128 threads (4 waves); macro-tile = 64 M x (NT*16) N; each wave
// owns one 16-row A tile + NT accumulators -> NT WMMAs per A staging pass.
//   OMODE: 0 = bf16 NHWC, 1 = f32 NHWC (z), 2 = f32 NCHW (x_recon)
// ---------------------------------------------------------------------------
template <int HIN, int WIN, int CIN, int HOUT, int WOUT, int COUT,
          int KH_, int KW_, int STR, int PAD_, int DIL_,
          bool IRELU, bool ORELU, int OMODE, int NT>
__global__ __launch_bounds__(128) void conv_wmma(
    const unsigned short* __restrict__ in, const float* __restrict__ wgt,
    const float* __restrict__ bias, const unsigned short* __restrict__ resid,
    void* __restrict__ out)
{
  constexpr int M    = 32 * HOUT * WOUT;
  constexpr int KTOT = KH_ * KW_ * CIN;

  __shared__ __align__(16) unsigned short As[4][16][32];       // per-wave A
  __shared__ __align__(16) unsigned short Bs[NT * 16][32];     // shared B [n][k]

  const int tid  = threadIdx.x;
  const int wave = tid >> 5;
  const int lane = tid & 31;
  const int hf   = lane >> 4;
  const int nl   = lane & 15;

  const int mtile  = blockIdx.x * 64 + wave * 16;
  const int ntile0 = blockIdx.y * (NT * 16);

  v8f acc[NT] = {};

  for (int kk = 0; kk < KTOT; kk += 32) {
    // ---- stage A (per wave): row i, column = lane (cin-contiguous) ----
    #pragma unroll 4
    for (int i = 0; i < 16; ++i) {
      int m = mtile + i;
      int k = kk + lane;
      unsigned short val = 0;
      if (k < KTOT) {
        int cin = k % CIN;
        int r   = k / CIN;
        int kx  = r % KW_;
        int ky  = r / KW_;
        int x   = m % WOUT;
        int r2  = m / WOUT;
        int y   = r2 % HOUT;
        int b   = r2 / HOUT;
        int iy = y * STR - PAD_ + ky;
        int ix = x * STR - PAD_ + kx;
        bool ok = true;
        if (DIL_ == 2) {                    // lhs-dilated (transpose conv)
          ok = ((iy & 1) == 0) && ((ix & 1) == 0);
          iy >>= 1; ix >>= 1;
        }
        if (ok && iy >= 0 && iy < HIN && ix >= 0 && ix < WIN) {
          unsigned short v = in[(((b * HIN + iy) * WIN) + ix) * CIN + cin];
          if (IRELU && (v & 0x8000u)) v = 0;
          val = v;
        }
      }
      As[wave][i][lane] = val;
    }
    // ---- stage B (block-wide): Bs[n][kcol] from OIHW fp32 weights ----
    #pragma unroll
    for (int i = 0; i < NT * 4; ++i) {
      int idx  = tid + i * 128;
      int n    = idx >> 5;
      int kcol = idx & 31;
      int k    = kk + kcol;
      unsigned short val = 0;
      int cout = ntile0 + n;
      if (cout < COUT && k < KTOT) {
        int cin = k % CIN;
        int r   = k / CIN;
        int kx  = r % KW_;
        int ky  = r / KW_;
        val = f2bf(wgt[(((cout * CIN) + cin) * KH_ + ky) * KW_ + kx]);
        if (i == 0 && kk + 32 < KTOT) {      // prefetch next K-step weights
          __builtin_prefetch(&wgt[cout * KTOT + kk + 32 + kcol], 0, 0);
        }
      }
      Bs[n][kcol] = val;
    }
    __syncthreads();
    // ---- fragments per ISA 16-bit A(16x32)/B(32x16) lane layouts ----
    v8u au;
    #pragma unroll
    for (int v = 0; v < 8; ++v) {
      int ka = ((v >> 2) << 4) + (hf << 3) + ((v & 3) << 1);
      au[v] = *(const unsigned int*)&As[wave][nl][ka];
    }
    v16bf a = __builtin_bit_cast(v16bf, au);
    #pragma unroll
    for (int t = 0; t < NT; ++t) {
      v8u bu;
      #pragma unroll
      for (int v = 0; v < 8; ++v) {
        int kb = ((v >> 2) << 4) + (hf << 3) + ((v & 3) << 1);
        bu[v] = *(const unsigned int*)&Bs[t * 16 + nl][kb];
      }
      v16bf b = __builtin_bit_cast(v16bf, bu);
      acc[t] = __builtin_amdgcn_wmma_f32_16x16x32_bf16(false, a, false, b,
                                                       (short)0, acc[t],
                                                       false, false);
    }
    __syncthreads();
  }

  // ---- epilogue: bias + residual + ReLU ----
  #pragma unroll
  for (int t = 0; t < NT; ++t) {
    const int ng = ntile0 + t * 16 + nl;
    if (ng < COUT) {
      #pragma unroll
      for (int v = 0; v < 8; ++v) {
        int m = mtile + v + (hf << 3);       // grid exact: m < M always
        float val = acc[t][v];
        if (bias)  val += bias[ng];
        if (resid) val += bf2f(resid[m * COUT + ng]);
        if (ORELU) val = fmaxf(val, 0.0f);
        if (OMODE == 0) {
          ((unsigned short*)out)[m * COUT + ng] = f2bf(val);
        } else if (OMODE == 1) {
          ((float*)out)[m * COUT + ng] = val;
        } else {
          int x = m % WOUT; int r2 = m / WOUT; int y = r2 % HOUT; int b = r2 / HOUT;
          ((float*)out)[(((b * COUT) + ng) * HOUT + y) * WOUT + x] = val;
        }
      }
    }
  }
}

// ---------------------------------------------------------------------------
// 1x1 conv (pure GEMM) with Tensor Data Mover A-tile staging.
// The A tile per 32-wide K step is a dense 2D tile: 16 rows x 32 bf16,
// row stride CIN, always fully in-bounds -> one wave-level TENSOR_LOAD_TO_LDS
// per K step replaces 16 per-lane gather+store rounds. Completion via
// TENSORcnt (s_wait_tensorcnt). Input ReLU (residual branch) is applied at
// fragment-build time since the DMA cannot transform data.
// Spatial size fixed at 32x32 (M = 32768) for all 1x1 layers in this net.
// ---------------------------------------------------------------------------
template <int CIN, int COUT, bool IRELU, bool ORELU, int OMODE, int NT>
__global__ __launch_bounds__(128) void conv1x1_wmma_tdm(
    const unsigned short* __restrict__ in, const float* __restrict__ wgt,
    const float* __restrict__ bias, const unsigned short* __restrict__ resid,
    void* __restrict__ out)
{
  __shared__ __align__(16) unsigned short As[4][16][32];
  __shared__ __align__(16) unsigned short Bs[NT * 16][32];

  const int tid  = threadIdx.x;
  const int wave = tid >> 5;
  const int lane = tid & 31;
  const int hf   = lane >> 4;
  const int nl   = lane & 15;

  const int mtile  = blockIdx.x * 64 + wave * 16;
  const int ntile0 = blockIdx.y * (NT * 16);

  // LDS byte offset of this wave's A tile (generic LDS pointers carry the
  // LDS offset in their low 32 bits per the flat-aperture mapping).
  const unsigned int laddr =
      (unsigned int)(unsigned long long)(const void*)&As[wave][0][0];

  v8f acc[NT] = {};

  for (int kk = 0; kk < CIN; kk += 32) {
    // ---- TDM descriptor: 16x32 bf16 tile, row stride CIN ----
    unsigned long long ga =
        (unsigned long long)(const void*)(in + (size_t)mtile * CIN + kk);
    v4u g0;
    g0[0] = 1u;                                          // count=1, user mode
    g0[1] = laddr;                                       // lds_addr
    g0[2] = (unsigned int)ga;                            // global_addr[31:0]
    g0[3] = ((unsigned int)(ga >> 32) & 0x01FFFFFFu)     // global_addr[56:32]
            | (2u << 30);                                // type=2 (image)
    v8u g1;
    g1[0] = 1u << 16;       // data_size=1 (2 bytes); no multicast/pad/iterate
    g1[1] = 32u << 16;      // tensor_dim0 = 32 (lo16 in hi half of dword1)
    g1[2] = 16u << 16;      // tensor_dim0 hi16 = 0 | tensor_dim1 lo16 = 16
    g1[3] = 32u << 16;      // tensor_dim1 hi16 = 0 | tile_dim0 = 32
    g1[4] = 16u;            // tile_dim1 = 16, tile_dim2 = 0
    g1[5] = (unsigned int)CIN;  // tensor_dim0_stride lo32 (data_size units)
    g1[6] = 0u;
    g1[7] = 0u;
    asm volatile("tensor_load_to_lds %0, %1" :: "s"(g0), "s"(g1) : "memory");

    // ---- stage B: contiguous w[cout][cin] rows, fp32 -> bf16 ----
    #pragma unroll
    for (int i = 0; i < NT * 4; ++i) {
      int idx  = tid + i * 128;
      int n    = idx >> 5;
      int kcol = idx & 31;
      int cout = ntile0 + n;
      Bs[n][kcol] = f2bf(wgt[cout * CIN + kk + kcol]);
      if (i == 0 && kk + 32 < CIN) {
        __builtin_prefetch(&wgt[cout * CIN + kk + 32 + kcol], 0, 0);
      }
    }
    __builtin_amdgcn_s_wait_tensorcnt(0);   // our wave's DMA landed in LDS
    __syncthreads();                        // Bs visible block-wide

    v8u au;
    #pragma unroll
    for (int v = 0; v < 8; ++v) {
      int ka = ((v >> 2) << 4) + (hf << 3) + ((v & 3) << 1);
      unsigned int u = *(const unsigned int*)&As[wave][nl][ka];
      if (IRELU) u = relu_pair(u);
      au[v] = u;
    }
    v16bf a = __builtin_bit_cast(v16bf, au);
    #pragma unroll
    for (int t = 0; t < NT; ++t) {
      v8u bu;
      #pragma unroll
      for (int v = 0; v < 8; ++v) {
        int kb = ((v >> 2) << 4) + (hf << 3) + ((v & 3) << 1);
        bu[v] = *(const unsigned int*)&Bs[t * 16 + nl][kb];
      }
      v16bf b = __builtin_bit_cast(v16bf, bu);
      acc[t] = __builtin_amdgcn_wmma_f32_16x16x32_bf16(false, a, false, b,
                                                       (short)0, acc[t],
                                                       false, false);
    }
    __syncthreads();
  }

  #pragma unroll
  for (int t = 0; t < NT; ++t) {
    const int ng = ntile0 + t * 16 + nl;
    if (ng < COUT) {
      #pragma unroll
      for (int v = 0; v < 8; ++v) {
        int m = mtile + v + (hf << 3);
        float val = acc[t][v];
        if (bias)  val += bias[ng];
        if (resid) val += bf2f(resid[m * COUT + ng]);
        if (ORELU) val = fmaxf(val, 0.0f);
        if (OMODE == 0) ((unsigned short*)out)[m * COUT + ng] = f2bf(val);
        else            ((float*)out)[m * COUT + ng] = val;
      }
    }
  }
}

// ---------------------------------------------------------------------------
// VQ argmin: score[m][n] = ||c_n||^2 - 2 * z_m . c_n  via f32 WMMA 16x16x4.
// ---------------------------------------------------------------------------
__global__ __launch_bounds__(128) void vq_argmin(
    const float* __restrict__ z, const float* __restrict__ cb,
    const float* __restrict__ cnorm, int* __restrict__ outidx)
{
  const int tid  = threadIdx.x;
  const int wave = tid >> 5;
  const int lane = tid & 31;
  const int hf   = lane >> 4;
  const int nl   = lane & 15;
  const int mtile = (blockIdx.x * 4 + wave) * 16;

  // f32 A(16x4): lanes0-15 hold K=0,1; lanes16-31 hold K=2,3 (M = lane&15)
  v2f af[16];
  {
    const float* zrow = z + (size_t)(mtile + nl) * 64 + hf * 2;
    #pragma unroll
    for (int j = 0; j < 16; ++j) { af[j][0] = zrow[4 * j]; af[j][1] = zrow[4 * j + 1]; }
  }

  float bestS[8]; int bestI[8];
  #pragma unroll
  for (int v = 0; v < 8; ++v) { bestS[v] = 3.4e38f; bestI[v] = 0; }

  for (int nt = 0; nt < 512; nt += 16) {
    v8f acc = {};
    const float* crow = cb + (size_t)(nt + nl) * 64 + hf * 2;
    #pragma unroll
    for (int j = 0; j < 16; ++j) {
      v2f b; b[0] = crow[4 * j]; b[1] = crow[4 * j + 1];
      acc = __builtin_amdgcn_wmma_f32_16x16x4_f32(false, af[j], false, b,
                                                  (short)0, acc, false, false);
    }
    float cn = cnorm[nt + nl];
    #pragma unroll
    for (int v = 0; v < 8; ++v) {
      float s = cn - 2.0f * acc[v];
      int  id = nt + nl;
      #pragma unroll
      for (int off = 8; off >= 1; off >>= 1) {   // reduce within 16-lane half
        float os = __shfl_xor(s, off, 32);
        int  oid = __shfl_xor(id, off, 32);
        if (os < s || (os == s && oid < id)) { s = os; id = oid; }
      }
      if (s < bestS[v] || (s == bestS[v] && id < bestI[v])) {
        bestS[v] = s; bestI[v] = id;
      }
    }
  }
  if (nl == 0) {
    #pragma unroll
    for (int v = 0; v < 8; ++v) outidx[mtile + v + (hf << 3)] = bestI[v];
  }
}

// ---------------------------------------------------------------------------
// Small helper kernels
// ---------------------------------------------------------------------------
__global__ void cnorm_kernel(const float* __restrict__ cb, float* __restrict__ cn) {
  int k = blockIdx.x * blockDim.x + threadIdx.x;
  if (k < 512) {
    float s = 0.0f;
    #pragma unroll 8
    for (int d = 0; d < 64; ++d) { float v = cb[k * 64 + d]; s += v * v; }
    cn[k] = s;
  }
}

__global__ void zero_kernel(unsigned int* __restrict__ p, int n) {
  int i = blockIdx.x * blockDim.x + threadIdx.x;
  if (i < n) p[i] = 0u;
}

__global__ void nchw_to_nhwc_bf16(const float* __restrict__ x,
                                  unsigned short* __restrict__ xb) {
  // B=32, C=3, H=W=128
  int t = blockIdx.x * blockDim.x + threadIdx.x;
  if (t < 32 * 3 * 128 * 128) {
    int c = t % 3; int r = t / 3; int xx = r & 127; r >>= 7; int y = r & 127; int b = r >> 7;
    xb[t] = f2bf(x[(((b * 3) + c) << 14) + (y << 7) + xx]);
  }
}

// Gather q = codebook[idx], write bf16 q (straight-through -> decoder input),
// accumulate sum((q - z)^2) and the code histogram.
__global__ __launch_bounds__(256) void vq_gather(
    const float* __restrict__ z, const float* __restrict__ cb,
    const int* __restrict__ idx, unsigned short* __restrict__ qb,
    unsigned int* __restrict__ counts, float* __restrict__ sqsum, int N)
{
  __shared__ float red[256];
  int m = blockIdx.x * blockDim.x + threadIdx.x;
  float s = 0.0f;
  if (m < N) {
    int id = idx[m];
    atomicAdd(&counts[id], 1u);
    const float* c  = cb + (size_t)id * 64;
    const float* zr = z + (size_t)m * 64;
    #pragma unroll 8
    for (int d = 0; d < 64; ++d) {
      float q = c[d];
      float df = q - zr[d];
      s += df * df;
      qb[(size_t)m * 64 + d] = f2bf(q);
    }
  }
  red[threadIdx.x] = s;
  __syncthreads();
  for (int o = 128; o > 0; o >>= 1) {
    if (threadIdx.x < o) red[threadIdx.x] += red[threadIdx.x + o];
    __syncthreads();
  }
  if (threadIdx.x == 0) atomicAdd(sqsum, red[0]);
}

// loss = 1.25 * mean((q - z)^2)  [q_latent == e_latent numerically]
// perplexity = exp(-sum p log(p + 1e-10)), p = counts / N
__global__ __launch_bounds__(512) void finalize_kernel(
    const unsigned int* __restrict__ counts, const float* __restrict__ sqsum,
    float* __restrict__ out_loss, float* __restrict__ out_perp,
    int N, float inv_elems)
{
  __shared__ float red[512];
  int k = threadIdx.x;
  float p = (float)counts[k] / (float)N;
  red[k] = p * logf(p + 1e-10f);
  __syncthreads();
  for (int o = 256; o > 0; o >>= 1) {
    if (k < o) red[k] += red[k + o];
    __syncthreads();
  }
  if (k == 0) {
    *out_perp = expf(-red[0]);
    *out_loss = 1.25f * (*sqsum) * inv_elems;
  }
}

// ---------------------------------------------------------------------------
// Launch
// ---------------------------------------------------------------------------
extern "C" void kernel_launch(void* const* d_in, const int* in_sizes, int n_in,
                              void* d_out, int out_size, void* d_ws, size_t ws_size,
                              hipStream_t stream) {
  (void)in_sizes; (void)n_in; (void)out_size; (void)ws_size;
  const float* x     = (const float*)d_in[0];
  const float* e_w1  = (const float*)d_in[1];  const float* e_b1 = (const float*)d_in[2];
  const float* e_w2  = (const float*)d_in[3];  const float* e_b2 = (const float*)d_in[4];
  const float* e_w3  = (const float*)d_in[5];  const float* e_b3 = (const float*)d_in[6];
  const float* e_r1a = (const float*)d_in[7];  const float* e_r1b = (const float*)d_in[8];
  const float* e_r2a = (const float*)d_in[9];  const float* e_r2b = (const float*)d_in[10];
  const float* pre_w = (const float*)d_in[11]; const float* pre_b = (const float*)d_in[12];
  const float* cb    = (const float*)d_in[13];
  const float* d_w1  = (const float*)d_in[14]; const float* d_b1 = (const float*)d_in[15];
  const float* d_r1a = (const float*)d_in[16]; const float* d_r1b = (const float*)d_in[17];
  const float* d_r2a = (const float*)d_in[18]; const float* d_r2b = (const float*)d_in[19];
  const float* dt_w1 = (const float*)d_in[20]; const float* dt_b1 = (const float*)d_in[21];
  const float* dt_w2 = (const float*)d_in[22]; const float* dt_b2 = (const float*)d_in[23];
  float* out = (float*)d_out;

  // --- workspace layout ---
  char* ws = (char*)d_ws;
  size_t off = 0;
  auto alloc = [&](size_t bytes) -> void* {
    void* p = ws + off; off += (bytes + 255) & ~(size_t)255; return p;
  };
  unsigned short* xb    = (unsigned short*)alloc((size_t)1572864 * 2); // x NHWC bf16
  unsigned short* big16 = (unsigned short*)alloc((size_t)8388608 * 2); // h1 / g4
  unsigned short* a8    = (unsigned short*)alloc((size_t)4194304 * 2);
  unsigned short* b8    = (unsigned short*)alloc((size_t)4194304 * 2);
  unsigned short* c8    = (unsigned short*)alloc((size_t)4194304 * 2);
  unsigned short* t32   = (unsigned short*)alloc((size_t)1048576 * 2); // res temps
  float*          zf    = (float*)alloc((size_t)2097152 * 4);          // z fp32 NHWC
  unsigned short* qb    = (unsigned short*)alloc((size_t)2097152 * 2); // q bf16 NHWC
  float*          cnrm  = (float*)alloc(512 * 4);
  int*            idxb  = (int*)alloc(32768 * 4);
  unsigned int*   cnts  = (unsigned int*)alloc(512 * 4);
  float*          sqsum = (float*)alloc(256);

  const int Nvq = 32768;  // 32 * 32 * 32 latent positions

  // zero scratch accumulators
  zero_kernel<<<4, 256, 0, stream>>>(cnts, 512);
  zero_kernel<<<1, 64, 0, stream>>>((unsigned int*)sqsum, 64);

  // x NCHW f32 -> NHWC bf16
  nchw_to_nhwc_bf16<<<(1572864 + 255) / 256, 256, 0, stream>>>(x, xb);

  // ----- encoder -----
  //             HIN  WIN  CIN  HOUT WOUT COUT KH KW S  P  D  IR    OR    OM NT
  conv_wmma<128, 128, 3,   64,  64,  64,  4, 4, 2, 1, 1, false, true, 0, 4>
      <<<dim3(2048, 1), 128, 0, stream>>>(xb, e_w1, e_b1, nullptr, big16);
  conv_wmma<64,  64,  64,  32,  32,  128, 4, 4, 2, 1, 1, false, true, 0, 4>
      <<<dim3(512, 2), 128, 0, stream>>>(big16, e_w2, e_b2, nullptr, a8);
  conv_wmma<32,  32,  128, 32,  32,  128, 3, 3, 1, 1, 1, false, false, 0, 4>
      <<<dim3(512, 2), 128, 0, stream>>>(a8, e_w3, e_b3, nullptr, b8);
  // res block 1 (1x1 second conv via TDM)
  conv_wmma<32,  32,  128, 32,  32,  32,  3, 3, 1, 1, 1, true,  false, 0, 2>
      <<<dim3(512, 1), 128, 0, stream>>>(b8, e_r1a, nullptr, nullptr, t32);
  conv1x1_wmma_tdm<32, 128, true, false, 0, 4>
      <<<dim3(512, 2), 128, 0, stream>>>(t32, e_r1b, nullptr, b8, c8);
  // res block 2 (+ final stack ReLU)
  conv_wmma<32,  32,  128, 32,  32,  32,  3, 3, 1, 1, 1, true,  false, 0, 2>
      <<<dim3(512, 1), 128, 0, stream>>>(c8, e_r2a, nullptr, nullptr, t32);
  conv1x1_wmma_tdm<32, 128, true, true, 0, 4>
      <<<dim3(512, 2), 128, 0, stream>>>(t32, e_r2b, nullptr, c8, a8);
  // pre-VQ 1x1 -> z (fp32 NHWC), TDM path
  conv1x1_wmma_tdm<128, 64, false, false, 1, 4>
      <<<dim3(512, 1), 128, 0, stream>>>(a8, pre_w, pre_b, nullptr, zf);

  // ----- vector quantizer -----
  cnorm_kernel<<<2, 256, 0, stream>>>(cb, cnrm);
  vq_argmin<<<Nvq / 64, 128, 0, stream>>>(zf, cb, cnrm, idxb);
  vq_gather<<<Nvq / 256, 256, 0, stream>>>(zf, cb, idxb, qb, cnts, sqsum, Nvq);
  finalize_kernel<<<1, 512, 0, stream>>>(cnts, sqsum, &out[0], &out[1 + 1572864],
                                         Nvq, 1.0f / (float)(Nvq * 64));

  // ----- decoder -----
  conv_wmma<32,  32,  64,  32,  32,  128, 3, 3, 1, 1, 1, false, false, 0, 4>
      <<<dim3(512, 2), 128, 0, stream>>>(qb, d_w1, d_b1, nullptr, b8);
  conv_wmma<32,  32,  128, 32,  32,  32,  3, 3, 1, 1, 1, true,  false, 0, 2>
      <<<dim3(512, 1), 128, 0, stream>>>(b8, d_r1a, nullptr, nullptr, t32);
  conv1x1_wmma_tdm<32, 128, true, false, 0, 4>
      <<<dim3(512, 2), 128, 0, stream>>>(t32, d_r1b, nullptr, b8, c8);
  conv_wmma<32,  32,  128, 32,  32,  32,  3, 3, 1, 1, 1, true,  false, 0, 2>
      <<<dim3(512, 1), 128, 0, stream>>>(c8, d_r2a, nullptr, nullptr, t32);
  conv1x1_wmma_tdm<32, 128, true, true, 0, 4>
      <<<dim3(512, 2), 128, 0, stream>>>(t32, d_r2b, nullptr, c8, a8);
  // ConvTranspose(4, s2, p1) == lhs-dilated (DIL=2) conv, pad 2
  conv_wmma<32,  32,  128, 64,  64,  64,  4, 4, 1, 2, 2, false, true,  0, 4>
      <<<dim3(2048, 1), 128, 0, stream>>>(a8, dt_w1, dt_b1, nullptr, big16);
  conv_wmma<64,  64,  64,  128, 128, 3,   4, 4, 1, 2, 2, false, false, 2, 1>
      <<<dim3(8192, 1), 128, 0, stream>>>(big16, dt_w2, dt_b2, nullptr, out + 1);
}
